// EncoderBlock_83184926589063
// MI455X (gfx1250) — matrix operand order repaired and checked by reading, not compiled, not run on hardware
//
#include <hip/hip_runtime.h>
#include <hip/hip_bf16.h>
#include <stdint.h>

#define SLEN   2048
#define DMODEL 1024
#define NHEAD  16
#define DHEAD  64
#define FFDIM  4096
#define BROWS  4096   // B * S

typedef __bf16 bf16_t;
typedef __bf16 v16bf __attribute__((ext_vector_type(16)));
typedef __bf16 v8bf  __attribute__((ext_vector_type(8)));
typedef float  v8f   __attribute__((ext_vector_type(8)));
typedef int    v4i_gcc __attribute__((vector_size(16)));   // matches builtin's V4i

#define AS_GLOBAL __attribute__((address_space(1)))
#define AS_LOCAL  __attribute__((address_space(3)))

static __device__ inline v16bf cat8(v8bf lo, v8bf hi) {
  return __builtin_shufflevector(lo, hi, 0,1,2,3,4,5,6,7,8,9,10,11,12,13,14,15);
}
static __device__ inline v8f vzero8() {
  v8f z;
#pragma unroll
  for (int i = 0; i < 8; ++i) z[i] = 0.0f;
  return z;
}
static __device__ inline v8f wmma_bf16(v16bf a, v16bf b, v8f c) {
  // D(f32 16x16) = A(16x32 bf16) * B(32x16 bf16) + C
  return __builtin_amdgcn_wmma_f32_16x16x32_bf16(false, a, false, b, (short)0, c,
                                                 false, false);
}

// Async global->LDS copy (CDNA5 GLOBAL_LOAD_ASYNC_TO_LDS_B128, ASYNCcnt-tracked).
static __device__ inline void async_copy_b128(const void* gp, void* lp) {
#if defined(__has_builtin) && __has_builtin(__builtin_amdgcn_global_load_async_to_lds_b128)
  __builtin_amdgcn_global_load_async_to_lds_b128(
      (AS_GLOBAL v4i_gcc*)(uintptr_t)gp,
      (AS_LOCAL v4i_gcc*)(unsigned)(uintptr_t)lp, 0, 0);
#else
  asm volatile("global_load_async_to_lds_b128 %0, %1, off"
               :: "v"((unsigned)(uintptr_t)lp), "v"(gp) : "memory");
#endif
}
static __device__ inline void wait_async0() {
#if defined(__has_builtin) && __has_builtin(__builtin_amdgcn_s_wait_asynccnt)
  __builtin_amdgcn_s_wait_asynccnt(0);
#else
  asm volatile("s_wait_asynccnt 0" ::: "memory");
#endif
}

// ---------------------------------------------------------------------------
// Weight transpose + f32 -> bf16 convert:  src[R][C] -> dst[C][R]
// ---------------------------------------------------------------------------
__global__ __launch_bounds__(256)
void transpose_to_bf16_kernel(const float* __restrict__ src, bf16_t* __restrict__ dst,
                              int R, int C) {
  __shared__ float tile[32][33];
  const int c0 = blockIdx.x * 32;
  const int r0 = blockIdx.y * 32;
  const int tx = threadIdx.x & 31;
  const int ty = threadIdx.x >> 5;   // 0..7
#pragma unroll
  for (int j = 0; j < 4; ++j)
    tile[ty + j * 8][tx] = src[(size_t)(r0 + ty + j * 8) * C + c0 + tx];
  __syncthreads();
#pragma unroll
  for (int j = 0; j < 4; ++j)
    dst[(size_t)(c0 + ty + j * 8) * R + r0 + tx] = (bf16_t)tile[tx][ty + j * 8];
}

__global__ __launch_bounds__(256)
void to_bf16_kernel(const float* __restrict__ src, bf16_t* __restrict__ dst, int n) {
  int i = blockIdx.x * 1024 + threadIdx.x;
#pragma unroll
  for (int j = 0; j < 4; ++j, i += 256)
    if (i < n) dst[i] = (bf16_t)src[i];
}

// ---------------------------------------------------------------------------
// Tiled WMMA GEMM:  C[M,N] = A[M,K](bf16) @ Bt[K,N](bf16) + bias
// 256 threads (8 waves), macro tile 128x128, k-step 32, double-buffered
// async global->LDS staging. Waves 4(M) x 2(N); wave tile 32x64 = 2x4 frags.
// ---------------------------------------------------------------------------
#define BM 128
#define BN 128
#define BK 32

template <bool STORE_F32, bool STORE_BF16>
__global__ __launch_bounds__(256, 2)
void gemm_bf16_kernel(const bf16_t* __restrict__ A, const bf16_t* __restrict__ Bt,
                      const float* __restrict__ bias,
                      float* __restrict__ outF, bf16_t* __restrict__ outB,
                      int M, int N, int K) {
  __shared__ bf16_t As[2][BM][BK + 8];   // +8 bf16 pad: 16B-aligned rows, fewer conflicts
  __shared__ bf16_t Bs[2][BK][BN + 8];
  const int tid   = threadIdx.x;
  const int wave  = tid >> 5;
  const int lane  = tid & 31;
  const int lrow  = lane & 15;
  const int lhalf = lane >> 4;
  const int wm = (wave >> 1) * 32;    // 4 waves along M
  const int wn = (wave & 1) * 64;     // 2 waves along N
  const int bm = blockIdx.y * BM;
  const int bn = blockIdx.x * BN;

  v8f acc[2][4];
#pragma unroll
  for (int i = 0; i < 2; ++i)
#pragma unroll
    for (int j = 0; j < 4; ++j) acc[i][j] = vzero8();

  const int ar = tid >> 1, ac = (tid & 1) * 16;   // A tile: 128 rows x 32 cols
  const int br = tid >> 3, bc = (tid & 7) * 16;   // B tile: 32 rows x 128 cols

  auto issue_tiles = [&](int k0, int buf) {
    const bf16_t* ag = A + (size_t)(bm + ar) * K + k0 + ac;
    async_copy_b128(ag,     &As[buf][ar][ac]);
    async_copy_b128(ag + 8, &As[buf][ar][ac + 8]);
    const bf16_t* bg = Bt + (size_t)(k0 + br) * N + bn + bc;
    async_copy_b128(bg,     &Bs[buf][br][bc]);
    async_copy_b128(bg + 8, &Bs[buf][br][bc + 8]);
  };

  issue_tiles(0, 0);
  int cur = 0;
  for (int k0 = 0; k0 < K; k0 += BK) {
    wait_async0();        // my async copies into As/Bs[cur] are done
    __syncthreads();      // everyone's copies into [cur] done; [cur^1] free
    if (k0 + BK < K) issue_tiles(k0 + BK, cur ^ 1);   // prefetch next tile

    v16bf afrag[2], bfrag[4];
#pragma unroll
    for (int mi = 0; mi < 2; ++mi) {
      // A layout: lane -> row M=lane%16; elems 0..7 = K 8*half+j, 8..15 = K 16+8*half+j
      const bf16_t* p = &As[cur][wm + mi * 16 + lrow][0];
      afrag[mi] = cat8(*(const v8bf*)(p + 8 * lhalf),
                       *(const v8bf*)(p + 16 + 8 * lhalf));
    }
#pragma unroll
    for (int ni = 0; ni < 4; ++ni) {
      // B layout: lane -> col N=lane%16; elems 0..7 = K 8*half+j, 8..15 = K 16+8*half+j
#pragma unroll
      for (int e = 0; e < 16; ++e)
        bfrag[ni][e] = Bs[cur][(e & 7) + 8 * lhalf + 16 * (e >> 3)][wn + ni * 16 + lrow];
    }
#pragma unroll
    for (int mi = 0; mi < 2; ++mi)
#pragma unroll
      for (int ni = 0; ni < 4; ++ni)
        acc[mi][ni] = wmma_bf16(afrag[mi], bfrag[ni], acc[mi][ni]);

    __syncthreads();      // all reads of [cur] complete before it becomes a target
    cur ^= 1;
  }

  // Epilogue: C layout vgpr r -> (M = r + 8*lhalf, N = lane%16); straight-line stores
#pragma unroll
  for (int ni = 0; ni < 4; ++ni) {
    const int n = bn + wn + ni * 16 + lrow;
    const float bv = bias[n];
#pragma unroll
    for (int mi = 0; mi < 2; ++mi) {
#pragma unroll
      for (int r = 0; r < 8; ++r) {
        const int m = bm + wm + mi * 16 + r + 8 * lhalf;
        const float v = acc[mi][ni][r] + bv;
        const size_t idx = (size_t)m * N + n;
        if constexpr (STORE_F32)  outF[idx] = v;
        if constexpr (STORE_BF16) outB[idx] = (bf16_t)v;
      }
    }
  }
}

// ---------------------------------------------------------------------------
// Flash attention (Q==K==V, shared qw tensor). One wave per 16-row q tile,
// kv tiles of 32 keys. scores = (Q Q^T)/8, masked online softmax, O = P Q.
// ---------------------------------------------------------------------------
__global__ __launch_bounds__(256, 2)
void attn_kernel(const bf16_t* __restrict__ Q, const int* __restrict__ mask,
                 bf16_t* __restrict__ Oattn) {
  __shared__ bf16_t ktT[8][DHEAD][40];  // per-wave K^T tile [d][key], padded rows
  __shared__ bf16_t pt[8][16][40];      // per-wave P tile   [m][key], padded rows
  const int tid   = threadIdx.x;
  const int wave  = tid >> 5;
  const int lane  = tid & 31;
  const int lrow  = lane & 15;
  const int lhalf = lane >> 4;
  const int tile  = blockIdx.x * 8 + wave;
  const int qt = tile & (SLEN / 16 - 1);        // 0..127
  const int h  = (tile >> 7) & (NHEAD - 1);     // 0..15
  const int b  = tile >> 11;                    // 0..1
  const bf16_t* Qbh = Q + (size_t)b * SLEN * DMODEL + h * DHEAD;
  const int q0 = qt * 16;

  // Q tile as two A fragments (d 0..31, 32..63)
  v16bf qa[2];
  {
    const bf16_t* qrow = Qbh + (size_t)(q0 + lrow) * DMODEL;
#pragma unroll
    for (int di = 0; di < 2; ++di)
      qa[di] = cat8(*(const v8bf*)(qrow + di * 32 + 8 * lhalf),
                    *(const v8bf*)(qrow + di * 32 + 16 + 8 * lhalf));
  }

  v8f oacc[4];
#pragma unroll
  for (int dq = 0; dq < 4; ++dq) oacc[dq] = vzero8();
  float mrow[8], lsum[8];
#pragma unroll
  for (int r = 0; r < 8; ++r) { mrow[r] = -1e30f; lsum[r] = 0.0f; }

  for (int kv = 0; kv < SLEN; kv += 32) {
    // Stage K-block transposed: lane loads key row (kv+lane), scatters to ktT[d][lane]
    {
      const bf16_t* krow = Qbh + (size_t)(kv + lane) * DMODEL;
#pragma unroll
      for (int c = 0; c < DHEAD; c += 8) {
        v8bf v = *(const v8bf*)(krow + c);
#pragma unroll
        for (int e = 0; e < 8; ++e) ktT[wave][c + e][lane] = v[e];
      }
    }
    asm volatile("s_wait_dscnt 0" ::: "memory");  // wave-private LDS RAW

    // scores: two 16x16 fragments (keys +0..15, +16..31); DK=64 -> 2 WMMAs each
    v8f sfrag[2];
#pragma unroll
    for (int nf = 0; nf < 2; ++nf) {
      v8f s = vzero8();
#pragma unroll
      for (int di = 0; di < 2; ++di) {
        v16bf kb;
#pragma unroll
        for (int e = 0; e < 16; ++e)
          kb[e] = ktT[wave][di * 32 + (e & 7) + 8 * lhalf + 16 * (e >> 3)][nf * 16 + lrow];
        s = wmma_bf16(qa[di], kb, s);
      }
      sfrag[nf] = s;
    }

    // scale, mask, online softmax (row lives across lanes 0..15 of each half)
    float corr[8];
#pragma unroll
    for (int r = 0; r < 8; ++r) {
      const int gq = q0 + r + 8 * lhalf;
      float s0 = sfrag[0][r] * 0.125f;
      float s1 = sfrag[1][r] * 0.125f;
      if (mask[(size_t)gq * SLEN + kv + lrow] == 0)      s0 = -1e9f;
      if (mask[(size_t)gq * SLEN + kv + 16 + lrow] == 0) s1 = -1e9f;
      float vmax = fmaxf(s0, s1);
#pragma unroll
      for (int off = 1; off < 16; off <<= 1)
        vmax = fmaxf(vmax, __shfl_xor(vmax, off, 32));
      const float mnew = fmaxf(mrow[r], vmax);
      const float c  = __expf(mrow[r] - mnew);
      const float p0 = __expf(s0 - mnew);
      const float p1 = __expf(s1 - mnew);
      float ps = p0 + p1;
#pragma unroll
      for (int off = 1; off < 16; off <<= 1)
        ps += __shfl_xor(ps, off, 32);
      lsum[r] = lsum[r] * c + ps;
      mrow[r] = mnew;
      corr[r] = c;
      sfrag[0][r] = p0;
      sfrag[1][r] = p1;
    }

#pragma unroll
    for (int dq = 0; dq < 4; ++dq)
#pragma unroll
      for (int r = 0; r < 8; ++r) oacc[dq][r] *= corr[r];

    // P: C layout -> LDS -> A layout (bf16)
#pragma unroll
    for (int nf = 0; nf < 2; ++nf)
#pragma unroll
      for (int r = 0; r < 8; ++r)
        pt[wave][r + 8 * lhalf][nf * 16 + lrow] = (bf16_t)sfrag[nf][r];
    asm volatile("s_wait_dscnt 0" ::: "memory");

    v16bf pa;
    {
      const bf16_t* p = &pt[wave][lrow][0];
      pa = cat8(*(const v8bf*)(p + 8 * lhalf), *(const v8bf*)(p + 16 + 8 * lhalf));
    }
    // O += P @ V   (V gathered from global; lane = d column, elems = key in table order)
#pragma unroll
    for (int dq = 0; dq < 4; ++dq) {
      v16bf vb;
#pragma unroll
      for (int e = 0; e < 16; ++e)
        vb[e] = Qbh[(size_t)(kv + (e & 7) + 8 * lhalf + 16 * (e >> 3)) * DMODEL + dq * 16 + lrow];
      oacc[dq] = wmma_bf16(pa, vb, oacc[dq]);
    }
  }

#pragma unroll
  for (int dq = 0; dq < 4; ++dq)
#pragma unroll
    for (int r = 0; r < 8; ++r) {
      const int gq = q0 + r + 8 * lhalf;
      const float ov = oacc[dq][r] / lsum[r];
      Oattn[((size_t)b * SLEN + gq) * DMODEL + h * DHEAD + dq * 16 + lrow] = (bf16_t)ov;
    }
}

// ---------------------------------------------------------------------------
// out = alpha*( (X+Y) - mean ) / (std_unbiased + 1e-8) + beta ; one row/block
// ---------------------------------------------------------------------------
template <bool WRITE_BF16>
__global__ __launch_bounds__(256)
void ln_residual_kernel(const float* __restrict__ X, const float* __restrict__ Y,
                        const float* __restrict__ alpha, const float* __restrict__ beta,
                        float* __restrict__ outF, bf16_t* __restrict__ outB) {
  const int row = blockIdx.x;
  const int tid = threadIdx.x;
  const float* x = X + (size_t)row * DMODEL;
  const float* y = Y + (size_t)row * DMODEL;
  __shared__ float red0[8], red1[8];
  float vals[4];
  float s = 0.0f, sq = 0.0f;
#pragma unroll
  for (int i = 0; i < 4; ++i) {
    const float v = x[tid + i * 256] + y[tid + i * 256];
    vals[i] = v; s += v; sq += v * v;
  }
#pragma unroll
  for (int off = 1; off < 32; off <<= 1) {
    s  += __shfl_xor(s, off, 32);
    sq += __shfl_xor(sq, off, 32);
  }
  if ((tid & 31) == 0) { red0[tid >> 5] = s; red1[tid >> 5] = sq; }
  __syncthreads();
  float ts = 0.0f, tq = 0.0f;
#pragma unroll
  for (int w = 0; w < 8; ++w) { ts += red0[w]; tq += red1[w]; }
  const float mean = ts * (1.0f / DMODEL);
  const float var  = fmaxf((tq - ts * mean) * (1.0f / (DMODEL - 1)), 0.0f);
  const float inv  = alpha[0] / (sqrtf(var) + 1e-8f);
  const float bet  = beta[0];
#pragma unroll
  for (int i = 0; i < 4; ++i) {
    const float o = (vals[i] - mean) * inv + bet;
    const size_t idx = (size_t)row * DMODEL + tid + i * 256;
    outF[idx] = o;
    if constexpr (WRITE_BF16) outB[idx] = (bf16_t)o;
  }
}

// ---------------------------------------------------------------------------
extern "C" void kernel_launch(void* const* d_in, const int* in_sizes, int n_in,
                              void* d_out, int out_size, void* d_ws, size_t ws_size,
                              hipStream_t stream) {
  const float* x      = (const float*)d_in[0];
  const int*   mask   = (const int*)d_in[1];
  const float* wq     = (const float*)d_in[2];
  const float* bq     = (const float*)d_in[3];
  const float* wo     = (const float*)d_in[4];
  const float* bo     = (const float*)d_in[5];
  const float* w1     = (const float*)d_in[6];
  const float* b1     = (const float*)d_in[7];
  const float* w2     = (const float*)d_in[8];
  const float* b2     = (const float*)d_in[9];
  const float* alpha1 = (const float*)d_in[10];
  const float* beta1  = (const float*)d_in[11];
  const float* alpha2 = (const float*)d_in[12];
  const float* beta2  = (const float*)d_in[13];
  float* out = (float*)d_out;
  (void)in_sizes; (void)n_in; (void)out_size; (void)ws_size;

  char* ws = (char*)d_ws;
  size_t off = 0;
  auto take = [&](size_t bytes) {
    char* p = ws + off;
    off += (bytes + 255) & ~(size_t)255;
    return p;
  };
  bf16_t* wqT   = (bf16_t*)take((size_t)DMODEL * DMODEL * 2);
  bf16_t* woT   = (bf16_t*)take((size_t)DMODEL * DMODEL * 2);
  bf16_t* w1T   = (bf16_t*)take((size_t)DMODEL * FFDIM * 2);
  bf16_t* w2T   = (bf16_t*)take((size_t)FFDIM * DMODEL * 2);
  bf16_t* xbf   = (bf16_t*)take((size_t)BROWS * DMODEL * 2);
  bf16_t* qbf   = (bf16_t*)take((size_t)BROWS * DMODEL * 2);
  bf16_t* attnb = (bf16_t*)take((size_t)BROWS * DMODEL * 2);
  float*  y2    = (float*) take((size_t)BROWS * DMODEL * 4);
  float*  x1    = (float*) take((size_t)BROWS * DMODEL * 4);
  bf16_t* x1bf  = (bf16_t*)take((size_t)BROWS * DMODEL * 2);
  bf16_t* hbf   = (bf16_t*)take((size_t)BROWS * FFDIM * 2);
  float*  y4    = (float*) take((size_t)BROWS * DMODEL * 4);

  dim3 blk(256);
  // weights -> bf16, transposed to [in][out]
  transpose_to_bf16_kernel<<<dim3(DMODEL / 32, DMODEL / 32), blk, 0, stream>>>(wq, wqT, DMODEL, DMODEL);
  transpose_to_bf16_kernel<<<dim3(DMODEL / 32, DMODEL / 32), blk, 0, stream>>>(wo, woT, DMODEL, DMODEL);
  transpose_to_bf16_kernel<<<dim3(DMODEL / 32, FFDIM / 32),  blk, 0, stream>>>(w1, w1T, FFDIM, DMODEL);
  transpose_to_bf16_kernel<<<dim3(FFDIM / 32, DMODEL / 32),  blk, 0, stream>>>(w2, w2T, DMODEL, FFDIM);
  to_bf16_kernel<<<dim3(BROWS * DMODEL / 1024), blk, 0, stream>>>(x, xbf, BROWS * DMODEL);

  // qw = x @ wq^T + bq  (bf16 out for attention)
  gemm_bf16_kernel<false, true><<<dim3(DMODEL / BN, BROWS / BM), blk, 0, stream>>>(
      xbf, wqT, bq, nullptr, qbf, BROWS, DMODEL, DMODEL);
  // flash attention over qw (Q==K==V)
  attn_kernel<<<dim3(2 * NHEAD * (SLEN / 16) / 8), blk, 0, stream>>>(qbf, mask, attnb);
  // attn_out = attn @ wo^T + bo  (f32 out)
  gemm_bf16_kernel<true, false><<<dim3(DMODEL / BN, BROWS / BM), blk, 0, stream>>>(
      attnb, woT, bo, y2, nullptr, BROWS, DMODEL, DMODEL);
  // x1 = LN(x + attn_out)
  ln_residual_kernel<true><<<dim3(BROWS), blk, 0, stream>>>(x, y2, alpha1, beta1, x1, x1bf);
  // h = x1 @ w1^T + b1  (bf16 out)
  gemm_bf16_kernel<false, true><<<dim3(FFDIM / BN, BROWS / BM), blk, 0, stream>>>(
      x1bf, w1T, b1, nullptr, hbf, BROWS, FFDIM, DMODEL);
  // ffn = h @ w2^T + b2  (f32 out)
  gemm_bf16_kernel<true, false><<<dim3(DMODEL / BN, BROWS / BM), blk, 0, stream>>>(
      hbf, w2T, b2, y4, nullptr, BROWS, DMODEL, FFDIM);
  // out = LN(x1 + ffn)
  ln_residual_kernel<false><<<dim3(BROWS), blk, 0, stream>>>(x1, y4, alpha2, beta2, out, nullptr);
}
